// BatchWhiten_16690242912792
// MI455X (gfx1250) — matrix operand order repaired and checked by reading, not compiled, not run on hardware
//
#include <hip/hip_runtime.h>
#include <math.h>

// ---------------------------------------------------------------------------
// BatchWhiten for MI455X (gfx1250, wave32, WMMA).
//   covar  = x^T x / N           (split-K bf16x3 WMMA + f32 atomics)
//   A      = max(0.1*running + 0.9*covar, 1e-5)
//   B      = A^{-1/2}            (Newton-Schulz iteration, WMMA GEMMs)
//   out    = x @ B               (bf16x3 WMMA GEMM)
// bf16x3: v = hi + lo (both bf16); v*w ~= hi*hi + hi*lo + lo*hi  (f32 acc)
// ---------------------------------------------------------------------------

typedef __attribute__((ext_vector_type(4)))  unsigned int   v4u;
typedef __attribute__((ext_vector_type(8)))  float          v8f;
typedef __attribute__((ext_vector_type(16))) unsigned short v16us;
typedef __attribute__((ext_vector_type(16))) __bf16         v16bf;

struct Pack32 { v4u a, b; };

static __device__ inline unsigned short f2bf_rne(float f) {
  unsigned int u = __float_as_uint(f);
  u += 0x7fffu + ((u >> 16) & 1u);
  return (unsigned short)(u >> 16);
}
static __device__ inline float bf2f(unsigned short h) {
  return __uint_as_float(((unsigned int)h) << 16);
}
static __device__ inline void split_bf16(float v, unsigned short& hi, unsigned short& lo) {
  hi = f2bf_rne(v);
  lo = f2bf_rne(v - bf2f(hi));
}

// A fragment (16x32 bf16, M x K): lane half hf = lane>>4, row M = lane&15.
// elems 0..7  -> K = 8*hf + i ; elems 8..15 -> K = 16 + 8*hf + (i-8).
// LDS layout: row = M (or N), 32 contiguous K elements -> two b128 loads.
static __device__ inline v16us fragA(const unsigned short* row, int hf) {
  Pack32 p;
  p.a = *(const v4u*)(row + 8 * hf);
  p.b = *(const v4u*)(row + 16 + 8 * hf);
  return __builtin_bit_cast(v16us, p);
}
// B fragment (32x16 bf16, K x N): lane half hf, col N = lane&15.
// elems 0..7  -> K = 8*hf + i ; elems 8..15 -> K = 16 + 8*hf + (i-8).
static __device__ inline v16us fragB(const unsigned short* row, int hf) {
  Pack32 p;
  p.a = *(const v4u*)(row + 8 * hf);
  p.b = *(const v4u*)(row + 16 + 8 * hf);
  return __builtin_bit_cast(v16us, p);
}

static __device__ inline v8f wmma_bf16_acc(v16us a, v16us b, v8f c) {
  return __builtin_amdgcn_wmma_f32_16x16x32_bf16(
      false, __builtin_bit_cast(v16bf, a),
      false, __builtin_bit_cast(v16bf, b),
      (short)0, c, false, false);
}

// ---------------------------------------------------------------------------
// Kernel 1: xtx += x^T x  over one K-chunk of 2048 rows, one 128x128 tile.
// grid = (16 tiles, 128 k-chunks); block = 128 threads = 4 waves (2x2 of 64x64)
// ---------------------------------------------------------------------------
__global__ __launch_bounds__(128) void covar_tn_kernel(const float* __restrict__ x,
                                                       float* __restrict__ xtx) {
  __shared__ __align__(16) unsigned short lAhi[128][32], lAlo[128][32];
  __shared__ __align__(16) unsigned short lBhi[128][32], lBlo[128][32];

  const int ti = blockIdx.x >> 2, tj = blockIdx.x & 3;
  const int i0 = ti * 128, j0 = tj * 128;
  const size_t k0 = (size_t)blockIdx.y * 2048;

  const int tid = threadIdx.x, lane = tid & 31, wv = tid >> 5;
  const int wm0 = (wv >> 1) * 64, wn0 = (wv & 1) * 64;
  const int hf = lane >> 4, mr = lane & 15;

  v8f acc[4][4];
  for (int a = 0; a < 4; ++a)
    for (int b = 0; b < 4; ++b)
      for (int e = 0; e < 8; ++e) acc[a][b][e] = 0.0f;

  for (int ks = 0; ks < 2048; ks += 32) {
    // stage 32(K) x 128 slabs of x for A (cols i0..) and B (cols j0..),
    // transposed into LDS as [col][k] with on-the-fly bf16 hi/lo split.
    for (int t = 0; t < 8; ++t) {
      const int q = tid + (t << 7);          // 0..1023
      const int r = q >> 5;                  // k-local 0..31
      const int c = (q & 31) << 2;           // col-local 0..124
      const size_t rowoff = (k0 + (size_t)ks + (size_t)r) * 512;
      unsigned short h, l;
      const float4 av = *(const float4*)(x + rowoff + i0 + c);
      split_bf16(av.x, h, l); lAhi[c + 0][r] = h; lAlo[c + 0][r] = l;
      split_bf16(av.y, h, l); lAhi[c + 1][r] = h; lAlo[c + 1][r] = l;
      split_bf16(av.z, h, l); lAhi[c + 2][r] = h; lAlo[c + 2][r] = l;
      split_bf16(av.w, h, l); lAhi[c + 3][r] = h; lAlo[c + 3][r] = l;
      const float4 bv = *(const float4*)(x + rowoff + j0 + c);
      split_bf16(bv.x, h, l); lBhi[c + 0][r] = h; lBlo[c + 0][r] = l;
      split_bf16(bv.y, h, l); lBhi[c + 1][r] = h; lBlo[c + 1][r] = l;
      split_bf16(bv.z, h, l); lBhi[c + 2][r] = h; lBlo[c + 2][r] = l;
      split_bf16(bv.w, h, l); lBhi[c + 3][r] = h; lBlo[c + 3][r] = l;
    }
    __syncthreads();

    v16us ah[4], al[4];
    for (int tm = 0; tm < 4; ++tm) {
      const int r = wm0 + tm * 16 + mr;
      ah[tm] = fragA(&lAhi[r][0], hf);
      al[tm] = fragA(&lAlo[r][0], hf);
    }
    for (int tn = 0; tn < 4; ++tn) {
      const int r = wn0 + tn * 16 + mr;
      const v16us bh = fragB(&lBhi[r][0], hf);
      const v16us bl = fragB(&lBlo[r][0], hf);
      for (int tm = 0; tm < 4; ++tm) {
        acc[tm][tn] = wmma_bf16_acc(ah[tm], bh, acc[tm][tn]);
        acc[tm][tn] = wmma_bf16_acc(ah[tm], bl, acc[tm][tn]);
        acc[tm][tn] = wmma_bf16_acc(al[tm], bh, acc[tm][tn]);
      }
    }
    __syncthreads();
  }

  for (int tm = 0; tm < 4; ++tm)
    for (int tn = 0; tn < 4; ++tn)
      for (int v = 0; v < 8; ++v) {
        const int gm = i0 + wm0 + tm * 16 + v + 8 * hf;
        const int gn = j0 + wn0 + tn * 16 + mr;
        atomicAdd(&xtx[(size_t)gm * 512 + gn], acc[tm][tn][v]);
      }
}

// ---------------------------------------------------------------------------
// Kernel 2: C[M x 512] = effAlpha * (A[M x 512] @ B[512 x 512]) + gamma * I
// effAlpha = alpha * (*scale_ptr) if scale_ptr, else alpha. bf16x3 precision.
// Template tile: BM x BN per WG, WR x WC waves of (BM/WR) x (BN/WC).
// ---------------------------------------------------------------------------
template <int BM, int BN, int WR, int WC>
__global__ __launch_bounds__(WR* WC * 32) void gemm_nn_kernel(
    const float* __restrict__ A, const float* __restrict__ Bm, float* __restrict__ C,
    const float* __restrict__ scale_ptr, float alpha, float gamma) {
  constexpr int WMt = BM / WR, WNt = BN / WC;
  constexpr int TM = WMt / 16, TN = WNt / 16;
  constexpr int T = WR * WC * 32;

  __shared__ __align__(16) unsigned short lAhi[BM][32], lAlo[BM][32];
  __shared__ __align__(16) unsigned short lBhi[BN][32], lBlo[BN][32];

  const int tid = threadIdx.x, lane = tid & 31, wv = tid >> 5;
  const int wm0 = (wv / WC) * WMt, wn0 = (wv % WC) * WNt;
  const int hf = lane >> 4, mr = lane & 15;
  const int m0 = blockIdx.x * BM, n0 = blockIdx.y * BN;

  v8f acc[TM][TN];
  for (int a = 0; a < TM; ++a)
    for (int b = 0; b < TN; ++b)
      for (int e = 0; e < 8; ++e) acc[a][b][e] = 0.0f;

  for (int ks = 0; ks < 512; ks += 32) {
    // stage A rows (K contiguous, matches global layout)
    for (int q = tid; q < BM * 8; q += T) {
      const int m = q >> 3, kc = (q & 7) << 2;
      const float4 v = *(const float4*)(A + (size_t)(m0 + m) * 512 + ks + kc);
      unsigned short h, l;
      split_bf16(v.x, h, l); lAhi[m][kc + 0] = h; lAlo[m][kc + 0] = l;
      split_bf16(v.y, h, l); lAhi[m][kc + 1] = h; lAlo[m][kc + 1] = l;
      split_bf16(v.z, h, l); lAhi[m][kc + 2] = h; lAlo[m][kc + 2] = l;
      split_bf16(v.w, h, l); lAhi[m][kc + 3] = h; lAlo[m][kc + 3] = l;
    }
    // stage B transposed into [n][k]
    for (int q = tid; q < BN * 8; q += T) {
      const int k = q / (BN / 4), nc = (q % (BN / 4)) << 2;
      const float4 v = *(const float4*)(Bm + (size_t)(ks + k) * 512 + n0 + nc);
      unsigned short h, l;
      split_bf16(v.x, h, l); lBhi[nc + 0][k] = h; lBlo[nc + 0][k] = l;
      split_bf16(v.y, h, l); lBhi[nc + 1][k] = h; lBlo[nc + 1][k] = l;
      split_bf16(v.z, h, l); lBhi[nc + 2][k] = h; lBlo[nc + 2][k] = l;
      split_bf16(v.w, h, l); lBhi[nc + 3][k] = h; lBlo[nc + 3][k] = l;
    }
    __syncthreads();

    v16us ah[TM], al[TM];
    for (int tm = 0; tm < TM; ++tm) {
      const int r = wm0 + tm * 16 + mr;
      ah[tm] = fragA(&lAhi[r][0], hf);
      al[tm] = fragA(&lAlo[r][0], hf);
    }
    for (int tn = 0; tn < TN; ++tn) {
      const int r = wn0 + tn * 16 + mr;
      const v16us bh = fragB(&lBhi[r][0], hf);
      const v16us bl = fragB(&lBlo[r][0], hf);
      for (int tm = 0; tm < TM; ++tm) {
        acc[tm][tn] = wmma_bf16_acc(ah[tm], bh, acc[tm][tn]);
        acc[tm][tn] = wmma_bf16_acc(ah[tm], bl, acc[tm][tn]);
        acc[tm][tn] = wmma_bf16_acc(al[tm], bh, acc[tm][tn]);
      }
    }
    __syncthreads();
  }

  const float eff = alpha * (scale_ptr ? scale_ptr[0] : 1.0f);
  for (int tm = 0; tm < TM; ++tm)
    for (int tn = 0; tn < TN; ++tn)
      for (int v = 0; v < 8; ++v) {
        const int gm = m0 + wm0 + tm * 16 + v + 8 * hf;
        const int gn = n0 + wn0 + tn * 16 + mr;
        float val = eff * acc[tm][tn][v];
        if (gamma != 0.0f && gm == gn) val += gamma;
        C[(size_t)gm * 512 + gn] = val;
      }
}

// ---------------------------------------------------------------------------
// Small elementwise / reduction kernels
// ---------------------------------------------------------------------------
__global__ void zero_f32_kernel(float* __restrict__ p, int n) {
  const int i = blockIdx.x * blockDim.x + threadIdx.x;
  if (i < n) p[i] = 0.0f;
}

__global__ void prep_covar_kernel(const float* __restrict__ xtx,
                                  const float* __restrict__ running,
                                  float* __restrict__ Amat,
                                  float* __restrict__ sumsq) {
  const int i = blockIdx.x * blockDim.x + threadIdx.x;  // 0..262143
  float v = 0.1f * running[i] + (0.9f / 262144.0f) * xtx[i];
  v = fmaxf(v, 1e-5f);
  Amat[i] = v;
  float sq = v * v;
  for (int o = 16; o > 0; o >>= 1) sq += __shfl_xor(sq, o, 32);
  if ((threadIdx.x & 31) == 0) atomicAdd(sumsq, sq);
}

__global__ void ns_scalars_kernel(const float* __restrict__ sumsq,
                                  float* __restrict__ scalars) {
  const float s = sqrtf(sumsq[0]);     // Frobenius norm of A
  scalars[0] = 1.0f / s;               // Y0 = A / s
  scalars[1] = rsqrtf(s);              // A^{-1/2} = Z_inf / sqrt(s)
}

__global__ void ns_init_kernel(const float* __restrict__ Amat,
                               const float* __restrict__ scalars,
                               float* __restrict__ Y, float* __restrict__ Z) {
  const int i = blockIdx.x * blockDim.x + threadIdx.x;  // 0..262143
  const int r = i >> 9, c = i & 511;
  Y[i] = Amat[i] * scalars[0];
  Z[i] = (r == c) ? 1.0f : 0.0f;
}

// ---------------------------------------------------------------------------
extern "C" void kernel_launch(void* const* d_in, const int* in_sizes, int n_in,
                              void* d_out, int out_size, void* d_ws, size_t ws_size,
                              hipStream_t stream) {
  (void)in_sizes; (void)n_in; (void)out_size; (void)ws_size;
  const float* x       = (const float*)d_in[0];   // [262144, 512]
  const float* running = (const float*)d_in[1];   // [512, 512]
  float* out = (float*)d_out;                     // [262144, 512]
  float* ws  = (float*)d_ws;

  const size_t DD = 512 * 512;
  float* xtx     = ws;                 // [512,512] accumulated x^T x
  float* sumsq   = ws + DD;            // [1]
  float* scalars = ws + DD + 1;        // [2] {1/s, 1/sqrt(s)}
  float* Amat    = ws + DD + 16;       // [512,512]
  float* Y0      = Amat + DD;
  float* Y1      = Y0 + DD;
  float* Z0      = Y1 + DD;
  float* Z1      = Z0 + DD;
  float* Tm      = Z1 + DD;            // total ws use ~7.1 MB

  // zero the atomic accumulators (xtx + sumsq)
  {
    const int n = (int)DD + 1;
    zero_f32_kernel<<<(n + 255) / 256, 256, 0, stream>>>(xtx, n);
  }

  // covariance: 16 output tiles x 128 split-K chunks of 2048 rows
  covar_tn_kernel<<<dim3(16, 128), 128, 0, stream>>>(x, xtx);

  // A = max(0.1*running + 0.9*xtx/N, eps); sumsq = ||A||_F^2
  prep_covar_kernel<<<1024, 256, 0, stream>>>(xtx, running, Amat, sumsq);
  ns_scalars_kernel<<<1, 1, 0, stream>>>(sumsq, scalars);
  ns_init_kernel<<<1024, 256, 0, stream>>>(Amat, scalars, Y0, Z0);

  // Newton-Schulz: T = 1.5 I - 0.5 Z Y ; Y <- Y T ; Z <- T Z
  float* Yc = Y0; float* Yn = Y1; float* Zc = Z0; float* Zn = Z1;
  for (int it = 0; it < 10; ++it) {
    gemm_nn_kernel<64, 64, 2, 2><<<dim3(8, 8), 128, 0, stream>>>(
        Zc, Yc, Tm, nullptr, -0.5f, 1.5f);
    gemm_nn_kernel<64, 64, 2, 2><<<dim3(8, 8), 128, 0, stream>>>(
        Yc, Tm, Yn, nullptr, 1.0f, 0.0f);
    gemm_nn_kernel<64, 64, 2, 2><<<dim3(8, 8), 128, 0, stream>>>(
        Tm, Zc, Zn, nullptr, 1.0f, 0.0f);
    float* t;
    t = Yc; Yc = Yn; Yn = t;
    t = Zc; Zc = Zn; Zn = t;
  }

  // out = x @ (Z / sqrt(s))
  gemm_nn_kernel<128, 256, 2, 4><<<dim3(2048, 2), 256, 0, stream>>>(
      x, Zc, out, scalars + 1, 1.0f, 0.0f);
}
